// Nneighbors_42013370089988
// MI455X (gfx1250) — matrix operand order repaired and checked by reading, not compiled, not run on hardware
//
#include <hip/hip_runtime.h>

#define NENT  200000
#define NREL  500
#define BATCH 64
#define KPAD  512
#define KTOP  15
#define KMAIN 480   // 15 full K-steps of 32; tail handled explicitly
#define NEG_INF (-__builtin_inff())

typedef __attribute__((ext_vector_type(16))) __bf16 v16bf;
typedef __attribute__((ext_vector_type(8)))  float  v8f;

struct __align__(16) U4 { unsigned x, y, z, w; };
struct __align__(16) F4 { float x, y, z, w; };

static __device__ __forceinline__ unsigned short bfbits(float f) {
  __bf16 h = (__bf16)f;
  return __builtin_bit_cast(unsigned short, h);
}

// ---------------------------------------------------------------------------
// Kernel 0: gather query rows, convert f32 -> bf16, zero-pad K 500 -> 512.
// ---------------------------------------------------------------------------
__global__ void qprep_kernel(const float* __restrict__ ev,
                             const int* __restrict__ qents,
                             unsigned short* __restrict__ qbf) {
  const int b = blockIdx.x;
  const long long row = qents[b];
  for (int k = threadIdx.x; k < KPAD; k += blockDim.x) {
    const float f = (k < NREL) ? ev[row * NREL + k] : 0.0f;
    qbf[b * KPAD + k] = bfbits(f);
  }
}

// Load 4 A fragments (one per M-tile) from LDS, then issue 4 WMMAs.
static __device__ __forceinline__ void kstep_wmma(
    const unsigned short* __restrict__ qlds, int k0, int col16, int khalf,
    const v16bf& bfrag, v8f acc[4]) {
  U4 a[8];
#pragma unroll
  for (int mt = 0; mt < 4; ++mt) {
    const int base = (mt * 16 + col16) * KPAD + k0 + khalf * 16;
    a[2 * mt + 0] = *reinterpret_cast<const U4*>(&qlds[base]);
    a[2 * mt + 1] = *reinterpret_cast<const U4*>(&qlds[base + 8]);
  }
#pragma unroll
  for (int mt = 0; mt < 4; ++mt) {
    union { U4 q[2]; v16bf v; } au;
    au.q[0] = a[2 * mt + 0];
    au.q[1] = a[2 * mt + 1];
    acc[mt] = __builtin_amdgcn_wmma_f32_16x16x32_bf16(
        false, au.v, false, bfrag, (short)0, acc[mt], false, false);
  }
}

// ---------------------------------------------------------------------------
// Kernel 1: sim = Q[64x500] x EV^T via V_WMMA_F32_16X16X32_BF16.
// 8 waves/WG; each wave owns 16 candidate columns and all 64 batch rows
// (4 accumulators), so EV is streamed from HBM exactly once.
// ---------------------------------------------------------------------------
__global__ __launch_bounds__(256) void sim_wmma_kernel(
    const float* __restrict__ ev,
    const unsigned short* __restrict__ qbf,
    const int* __restrict__ qrels,
    float* __restrict__ sim,
    unsigned long long* __restrict__ mask) {
  __shared__ __align__(16) unsigned short qlds[BATCH * KPAD];  // 64 KB

  const int tid = threadIdx.x;
  // Stage the bf16 query tile into LDS (256 B per thread, b128 copies).
  {
    const U4* __restrict__ src = reinterpret_cast<const U4*>(qbf);
    U4* dst = reinterpret_cast<U4*>(qlds);
#pragma unroll
    for (int i = 0; i < 16; ++i) dst[tid + 256 * i] = src[tid + 256 * i];
  }
  __syncthreads();

  const int wave  = tid >> 5;
  const int lane  = tid & 31;
  const int col16 = lane & 15;   // candidate column within wave tile (B: N)
  const int khalf = lane >> 4;   // lane-half selects K range per WMMA layout

  const long long candw = (long long)blockIdx.x * 128 + wave * 16 + col16;
  const long long cand  = (candw < NENT) ? candw : (NENT - 1);  // clamp loads
  const float* __restrict__ brow = ev + cand * NREL;

  v8f acc[4];
  {
    const v8f z = {0.f, 0.f, 0.f, 0.f, 0.f, 0.f, 0.f, 0.f};
#pragma unroll
    for (int mt = 0; mt < 4; ++mt) acc[mt] = z;
  }

  // ---- Main K loop: 15 guaranteed-full steps, no guards, no divergence.
  for (int k0 = 0; k0 < KMAIN; k0 += 32) {
    const int kb = k0 + khalf * 16;          // lane's 16 contiguous K values
    const F4* __restrict__ p = reinterpret_cast<const F4*>(brow + kb);
    const F4 f0 = p[0];
    const F4 f1 = p[1];
    const F4 f2 = p[2];
    const F4 f3 = p[3];
    v16bf bfrag;
    bfrag[0]  = (__bf16)f0.x; bfrag[1]  = (__bf16)f0.y;
    bfrag[2]  = (__bf16)f0.z; bfrag[3]  = (__bf16)f0.w;
    bfrag[4]  = (__bf16)f1.x; bfrag[5]  = (__bf16)f1.y;
    bfrag[6]  = (__bf16)f1.z; bfrag[7]  = (__bf16)f1.w;
    bfrag[8]  = (__bf16)f2.x; bfrag[9]  = (__bf16)f2.y;
    bfrag[10] = (__bf16)f2.z; bfrag[11] = (__bf16)f2.w;
    bfrag[12] = (__bf16)f3.x; bfrag[13] = (__bf16)f3.y;
    bfrag[14] = (__bf16)f3.z; bfrag[15] = (__bf16)f3.w;

    __builtin_prefetch(brow + kb + 32, 0, 1);  // speculative: safe past end

    kstep_wmma(qlds, k0, col16, khalf, bfrag, acc);
  }

  // ---- K tail (k0 = 480): khalf==0 covers K 480..495 (all valid);
  //      khalf==1 covers K 496..511 (first float4 valid, rest zero).
  {
    const int kb = KMAIN + khalf * 16;       // 480 or 496; p[0] in-bounds
    const F4* __restrict__ p = reinterpret_cast<const F4*>(brow + kb);
    const F4 f0 = p[0];
    v16bf bfrag;
    bfrag[0] = (__bf16)f0.x; bfrag[1] = (__bf16)f0.y;
    bfrag[2] = (__bf16)f0.z; bfrag[3] = (__bf16)f0.w;
    if (khalf == 0) {
#pragma unroll
      for (int j = 1; j < 4; ++j) {
        const F4 f = p[j];
        bfrag[4 * j + 0] = (__bf16)f.x;
        bfrag[4 * j + 1] = (__bf16)f.y;
        bfrag[4 * j + 2] = (__bf16)f.z;
        bfrag[4 * j + 3] = (__bf16)f.w;
      }
    } else {
#pragma unroll
      for (int e = 4; e < 16; ++e) bfrag[e] = (__bf16)0.0f;
    }
    kstep_wmma(qlds, KMAIN, col16, khalf, bfrag, acc);
  }

  if (candw < NENT) {
    // C layout: VGPR r, lane -> (M = 16*mt + 8*khalf + r, N = col16).
#pragma unroll
    for (int mt = 0; mt < 4; ++mt) {
#pragma unroll
      for (int r = 0; r < 8; ++r) {
        const int M = mt * 16 + khalf * 8 + r;
        sim[(long long)M * NENT + candw] = acc[mt][r];
      }
    }
    // Relation keep-mask: row is L0/L2-hot right after the K streaming.
    if (khalf == 0) {
      unsigned long long m = 0ull;
      for (int b = 0; b < BATCH; ++b) {
        const float rv = brow[qrels[b]];
        if (rv != 0.0f && rv == rv) m |= (1ull << b);
      }
      mask[candw] = m;
    }
  }
}

// ---------------------------------------------------------------------------
// Kernel 2: per batch row, top-15 of masked sim over 200k candidates.
// Register-resident sorted insert per thread, then LDS tree merge.
// ---------------------------------------------------------------------------
__global__ __launch_bounds__(256) void topk_kernel(
    const float* __restrict__ sim,
    const unsigned long long* __restrict__ mask,
    float* __restrict__ out_vals,
    float* __restrict__ out_idx) {
  const int b = blockIdx.x;
  const int tid = threadIdx.x;
  const float* __restrict__ srow = sim + (long long)b * NENT;

  float v[KTOP]; int ix[KTOP];
#pragma unroll
  for (int j = 0; j < KTOP; ++j) { v[j] = NEG_INF; ix[j] = -1; }

  for (int i = tid; i < NENT; i += 256) {
    const bool keep = (mask[i] >> b) & 1ull;
    const float mv = keep ? srow[i] : NEG_INF;
    if (mv > v[KTOP - 1]) {
      v[KTOP - 1] = mv; ix[KTOP - 1] = i;
#pragma unroll
      for (int j = KTOP - 1; j > 0; --j) {
        if (v[j] > v[j - 1]) {
          const float tv = v[j]; v[j] = v[j - 1]; v[j - 1] = tv;
          const int   ti = ix[j]; ix[j] = ix[j - 1]; ix[j - 1] = ti;
        }
      }
    }
  }

  __shared__ float sv[256 * KTOP];
  __shared__ int   si[256 * KTOP];
#pragma unroll
  for (int j = 0; j < KTOP; ++j) { sv[tid * KTOP + j] = v[j]; si[tid * KTOP + j] = ix[j]; }
  __syncthreads();

  for (int stride = 128; stride >= 1; stride >>= 1) {
    float mv[KTOP]; int mi[KTOP];
    if (tid < stride) {
      int pa = 0, pb = 0;
#pragma unroll
      for (int j = 0; j < KTOP; ++j) {
        const float va = (pa < KTOP) ? sv[tid * KTOP + pa] : NEG_INF;
        const float vb = (pb < KTOP) ? sv[(tid + stride) * KTOP + pb] : NEG_INF;
        if (va >= vb) { mv[j] = va; mi[j] = si[tid * KTOP + pa]; ++pa; }
        else          { mv[j] = vb; mi[j] = si[(tid + stride) * KTOP + pb]; ++pb; }
      }
    }
    __syncthreads();
    if (tid < stride) {
#pragma unroll
      for (int j = 0; j < KTOP; ++j) { sv[tid * KTOP + j] = mv[j]; si[tid * KTOP + j] = mi[j]; }
    }
    __syncthreads();
  }

  if (tid == 0) {
#pragma unroll
    for (int j = 0; j < KTOP; ++j) {
      out_vals[b * KTOP + j] = sv[j];
      out_idx[b * KTOP + j]  = (float)si[j];
    }
  }
}

// ---------------------------------------------------------------------------
extern "C" void kernel_launch(void* const* d_in, const int* in_sizes, int n_in,
                              void* d_out, int out_size, void* d_ws, size_t ws_size,
                              hipStream_t stream) {
  (void)in_sizes; (void)n_in; (void)out_size; (void)ws_size;
  const float* ev    = (const float*)d_in[0];
  const int*   qents = (const int*)d_in[1];
  const int*   qrels = (const int*)d_in[2];

  float* out      = (float*)d_out;
  float* simbuf   = out;                                  // [64 x 200000]
  float* out_vals = out + (long long)BATCH * NENT;        // [64 x 15]
  float* out_idx  = out_vals + BATCH * KTOP;              // [64 x 15]

  unsigned short* qbf = (unsigned short*)d_ws;            // 64 KB
  unsigned long long* mask =
      (unsigned long long*)((char*)d_ws + (size_t)BATCH * KPAD * sizeof(unsigned short));
  // scratch usage: 64 KB + 200000*8 B = ~1.66 MB

  qprep_kernel<<<BATCH, 128, 0, stream>>>(ev, qents, qbf);

  const int ngrid = (NENT + 127) / 128;  // 128 candidate columns per WG
  sim_wmma_kernel<<<ngrid, 256, 0, stream>>>(ev, qbf, qrels, simbuf, mask);

  topk_kernel<<<BATCH, 256, 0, stream>>>(simbuf, mask, out_vals, out_idx);
}
